// BiMambaForMaskedLM_87866440942160
// MI455X (gfx1250) — compile-verified, matrix operand order checked by reading, not run on hardware
//
#include <hip/hip_runtime.h>
#include <hip/hip_bf16.h>

// ---------------- model constants ----------------
#define LSEQ 2048
#define VOC  32000
#define DMOD 768
#define NLAY 2
#define DI   1536           // 2*DMOD
#define SST  16
#define KCONV 4
#define RRK  48
#define DBCN 80             // R + 2S
#define DBCPAD 128          // padded N for x_proj gemm
#define RPAD 64             // padded K for dt gemm

typedef __attribute__((ext_vector_type(16))) _Float16 v16h;
typedef __attribute__((ext_vector_type(8)))  _Float16 v8h;
typedef __attribute__((ext_vector_type(8)))  float    v8f;

__device__ __forceinline__ float silu_f(float v) { return v / (1.0f + __expf(-v)); }
__device__ __forceinline__ float softplus_f(float v) {
    return (v > 20.0f) ? v : log1pf(__expf(v));
}

// ---------------------------------------------------------------------------
// Generic WMMA GEMM:  C[M,N] (fp32, ldc) = A[M,K] (f16, lda) * Bt[N,K]^T (f16, ldb)
// Requires: M % 64 == 0, N % 128 == 0, K % 32 == 0.
// Block = 256 threads (8 waves, 2x4); each wave computes a 32x32 tile via
// 2x2 v_wmma_f32_16x16x32_f16 per 32-wide K step.
// ---------------------------------------------------------------------------
__global__ __launch_bounds__(256) void gemm_wmma_f16(
    const _Float16* __restrict__ A, int lda,
    const _Float16* __restrict__ Bt, int ldb,
    float* __restrict__ C, int ldc, int K)
{
    const int lane  = threadIdx.x & 31;
    const int wave  = threadIdx.x >> 5;
    const int waveM = wave >> 2;           // 0..1
    const int waveN = wave & 3;            // 0..3
    const int mBase = blockIdx.y * 64  + waveM * 32;
    const int nBase = blockIdx.x * 128 + waveN * 32;
    const int hiHalf = lane >> 4;          // 0 for lanes 0-15, 1 for 16-31
    const int r      = lane & 15;
    const int kOff   = hiHalf ? 8 : 0;     // per ISA 16-bit A layout

    const _Float16* a0p = A  + (size_t)(mBase + r)      * lda + kOff;
    const _Float16* a1p = A  + (size_t)(mBase + 16 + r) * lda + kOff;
    const _Float16* b0p = Bt + (size_t)(nBase + r)      * ldb + kOff;
    const _Float16* b1p = Bt + (size_t)(nBase + 16 + r) * ldb + kOff;

    v8f c00 = {}, c01 = {}, c10 = {}, c11 = {};

    for (int k0 = 0; k0 < K; k0 += 32) {
        v8h a0l = *(const v8h*)(a0p + k0);
        v8h a0h = *(const v8h*)(a0p + k0 + 16);
        v8h a1l = *(const v8h*)(a1p + k0);
        v8h a1h = *(const v8h*)(a1p + k0 + 16);
        v8h b0l = *(const v8h*)(b0p + k0);
        v8h b0h = *(const v8h*)(b0p + k0 + 16);
        v8h b1l = *(const v8h*)(b1p + k0);
        v8h b1h = *(const v8h*)(b1p + k0 + 16);

        v16h a0 = __builtin_shufflevector(a0l, a0h, 0,1,2,3,4,5,6,7,8,9,10,11,12,13,14,15);
        v16h a1 = __builtin_shufflevector(a1l, a1h, 0,1,2,3,4,5,6,7,8,9,10,11,12,13,14,15);
        v16h b0 = __builtin_shufflevector(b0l, b0h, 0,1,2,3,4,5,6,7,8,9,10,11,12,13,14,15);
        v16h b1 = __builtin_shufflevector(b1l, b1h, 0,1,2,3,4,5,6,7,8,9,10,11,12,13,14,15);

        c00 = __builtin_amdgcn_wmma_f32_16x16x32_f16(false, a0, false, b0, (short)0, c00, false, false);
        c01 = __builtin_amdgcn_wmma_f32_16x16x32_f16(false, a0, false, b1, (short)0, c01, false, false);
        c10 = __builtin_amdgcn_wmma_f32_16x16x32_f16(false, a1, false, b0, (short)0, c10, false, false);
        c11 = __builtin_amdgcn_wmma_f32_16x16x32_f16(false, a1, false, b1, (short)0, c11, false, false);
    }

    // C/D layout: VGPR i -> row = i (lanes 0-15) or 8+i (lanes 16-31), col = lane&15
    const int rowAdd = hiHalf ? 8 : 0;
    {
        float* p = C + (size_t)(mBase + rowAdd) * ldc + nBase + r;
        #pragma unroll
        for (int i = 0; i < 8; i++) p[(size_t)i * ldc] = c00[i];
    }
    {
        float* p = C + (size_t)(mBase + rowAdd) * ldc + nBase + 16 + r;
        #pragma unroll
        for (int i = 0; i < 8; i++) p[(size_t)i * ldc] = c01[i];
    }
    {
        float* p = C + (size_t)(mBase + 16 + rowAdd) * ldc + nBase + r;
        #pragma unroll
        for (int i = 0; i < 8; i++) p[(size_t)i * ldc] = c10[i];
    }
    {
        float* p = C + (size_t)(mBase + 16 + rowAdd) * ldc + nBase + 16 + r;
        #pragma unroll
        for (int i = 0; i < 8; i++) p[(size_t)i * ldc] = c11[i];
    }
}

// ---------------------------------------------------------------------------
// Elementwise / helper kernels
// ---------------------------------------------------------------------------
__global__ __launch_bounds__(256) void f32_to_f16_kernel(
    const float* __restrict__ in, _Float16* __restrict__ out, int n)
{
    for (int i = blockIdx.x * 256 + threadIdx.x; i < n; i += gridDim.x * 256)
        out[i] = (_Float16)in[i];
}

// gather embeddings; also write the time-flipped copy for the backward branch
__global__ __launch_bounds__(256) void gather_emb_kernel(
    const int* __restrict__ ids, const float* __restrict__ emb,
    float* __restrict__ outF, float* __restrict__ outB)
{
    int l = blockIdx.x;
    int id = ids[l];
    #pragma unroll
    for (int k = 0; k < 3; k++) {
        int c = threadIdx.x + k * 256;
        float v = emb[(size_t)id * DMOD + c];
        outF[(size_t)l * DMOD + c] = v;
        outB[(size_t)(LSEQ - 1 - l) * DMOD + c] = v;
    }
}

// x_proj (80 x DI) -> padded f16 (128 x DI), zero rows 80..127
__global__ __launch_bounds__(256) void pad_xproj_kernel(
    const float* __restrict__ xp, _Float16* __restrict__ out)
{
    int i = blockIdx.x * 256 + threadIdx.x;           // over 128*DI
    int e = i / DI;
    out[i] = (e < DBCN) ? (_Float16)xp[(size_t)e * DI + (i - e * DI)] : (_Float16)0.0f;
}

// dt_w (DI x 48) -> padded f16 (DI x 64), zero cols 48..63
__global__ __launch_bounds__(256) void pad_dtw_kernel(
    const float* __restrict__ w, _Float16* __restrict__ out)
{
    int i = blockIdx.x * 256 + threadIdx.x;           // over DI*64
    int d = i >> 6, rr = i & 63;
    out[i] = (rr < RRK) ? (_Float16)w[(size_t)d * RRK + rr] : (_Float16)0.0f;
}

// dbc fp32 [L,128] -> f16 [L,64] (first 48 cols, rest zero) for dt gemm
__global__ __launch_bounds__(256) void make_dbc16_kernel(
    const float* __restrict__ dbc, _Float16* __restrict__ out)
{
    int i = blockIdx.x * 256 + threadIdx.x;           // over L*64
    int l = i >> 6, c = i & 63;
    out[i] = (c < RRK) ? (_Float16)dbc[(size_t)l * DBCPAD + c] : (_Float16)0.0f;
}

// residual update + RMSNorm -> f16 (optionally flipped rows / column offset for concat)
__global__ __launch_bounds__(256) void resnorm_kernel(
    const float* __restrict__ h, const float* __restrict__ res_in,
    float* __restrict__ res_out, const float* __restrict__ w,
    _Float16* __restrict__ out16, int ldout, int colOff, int rowFlip)
{
    int l = blockIdx.x, tid = threadIdx.x;
    __shared__ float red[256];
    float vals[3]; float ss = 0.0f;
    #pragma unroll
    for (int k = 0; k < 3; k++) {
        int c = tid + k * 256;
        float v = h[(size_t)l * DMOD + c];
        if (res_in) v += res_in[(size_t)l * DMOD + c];
        vals[k] = v; ss += v * v;
    }
    red[tid] = ss; __syncthreads();
    for (int s = 128; s > 0; s >>= 1) { if (tid < s) red[tid] += red[tid + s]; __syncthreads(); }
    float rn = rsqrtf(red[0] * (1.0f / DMOD) + 1e-5f);
    int lo = rowFlip ? (LSEQ - 1 - l) : l;
    #pragma unroll
    for (int k = 0; k < 3; k++) {
        int c = tid + k * 256;
        if (res_out) res_out[(size_t)l * DMOD + c] = vals[k];
        out16[(size_t)lo * ldout + colOff + c] = (_Float16)(vals[k] * rn * w[c]);
    }
}

// circular depthwise conv (K=4) + SiLU over x part of xz [L, 2*DI]
__global__ __launch_bounds__(256) void conv_silu_kernel(
    const float* __restrict__ xz, const float* __restrict__ cw,
    const float* __restrict__ cb, _Float16* __restrict__ out16)
{
    int idx = blockIdx.x * 256 + threadIdx.x;         // over L*DI
    int l = idx / DI, d = idx - l * DI;
    float acc = cb[d];
    #pragma unroll
    for (int k = 0; k < KCONV; k++) {
        int lk = l + k - (KCONV - 1);
        lk += (lk < 0) ? LSEQ : 0;
        acc += cw[d * KCONV + k] * xz[(size_t)lk * (2 * DI) + d];
    }
    out16[idx] = (_Float16)silu_f(acc);
}

// selective scan: thread = channel d; state [16] in registers across 2048 steps.
// B/C vectors (shared across channels) staged through LDS in 16-step chunks.
__global__ __launch_bounds__(256) void scan_kernel(
    const float* __restrict__ dtlin, const float* __restrict__ dbc,
    const _Float16* __restrict__ u16, const float* __restrict__ xz,
    const float* __restrict__ A_log, const float* __restrict__ Dp,
    const float* __restrict__ dt_b, _Float16* __restrict__ y16)
{
    int d = blockIdx.x * 256 + threadIdx.x;           // < DI
    float Av[SST];
    #pragma unroll
    for (int s = 0; s < SST; s++) Av[s] = -__expf(A_log[(size_t)d * SST + s]);
    float hs[SST];
    #pragma unroll
    for (int s = 0; s < SST; s++) hs[s] = 0.0f;
    float dpv = Dp[d], dtb = dt_b[d];

    __shared__ float sB[16 * SST];
    __shared__ float sC[16 * SST];

    for (int l0 = 0; l0 < LSEQ; l0 += 16) {
        __syncthreads();
        for (int i = threadIdx.x; i < 16 * 32; i += 256) {
            int rr = i >> 5, cc = i & 31;
            float v = dbc[(size_t)(l0 + rr) * DBCPAD + RRK + cc];
            if (cc < SST) sB[rr * SST + cc] = v;
            else          sC[rr * SST + (cc - SST)] = v;
        }
        __syncthreads();
        #pragma unroll 4
        for (int rr = 0; rr < 16; rr++) {
            int l = l0 + rr;
            float dt = softplus_f(dtlin[(size_t)l * DI + d] + dtb);
            float u  = (float)u16[(size_t)l * DI + d];
            float du = dt * u;
            float ys = 0.0f;
            #pragma unroll
            for (int s = 0; s < SST; s++) {
                float e = __expf(dt * Av[s]);
                hs[s] = e * hs[s] + du * sB[rr * SST + s];
                ys += hs[s] * sC[rr * SST + s];
            }
            float z = xz[(size_t)l * (2 * DI) + DI + d];
            y16[(size_t)l * DI + d] = (_Float16)((ys + dpv * u) * silu_f(z));
        }
    }
}

// ---------------------------------------------------------------------------
// Host-side launch
// ---------------------------------------------------------------------------
extern "C" void kernel_launch(void* const* d_in, const int* in_sizes, int n_in,
                              void* d_out, int out_size, void* d_ws, size_t ws_size,
                              hipStream_t stream)
{
    const int* ids           = (const int*)d_in[0];
    const float* embedding   = (const float*)d_in[1];
    const float* lm_head     = (const float*)d_in[2];
    // per-direction param tables (order matches _PNAMES)
    const float* P[2][11];
    for (int j = 0; j < 11; j++) {
        P[0][j] = (const float*)d_in[3 + j];
        P[1][j] = (const float*)d_in[14 + j];
    }
    enum { IN_PROJ, CONV_W, CONV_B, X_PROJ, DT_W, DT_B, A_LOG, DP, OUT_PROJ, NORM_W, NORM_F };

    // ---- workspace layout (256B aligned bump allocator) ----
    char* ws = (char*)d_ws;
    size_t off = 0;
    auto alloc = [&](size_t bytes) -> char* {
        char* p = ws + off;
        off = (off + bytes + 255) & ~(size_t)255;
        return p;
    };
    float*    embF      = (float*)alloc((size_t)LSEQ * DMOD * 4);
    float*    embB      = (float*)alloc((size_t)LSEQ * DMOD * 4);
    _Float16* emb16     = (_Float16*)alloc((size_t)VOC * DMOD * 2);
    _Float16* lmhead16  = (_Float16*)alloc((size_t)DMOD * DI * 2);
    _Float16* inproj16  = (_Float16*)alloc((size_t)2 * DI * DMOD * 2);
    _Float16* xproj16   = (_Float16*)alloc((size_t)DBCPAD * DI * 2);
    _Float16* dtw16     = (_Float16*)alloc((size_t)DI * RPAD * 2);
    _Float16* outproj16 = (_Float16*)alloc((size_t)DMOD * DI * 2);
    float*    res32     = (float*)alloc((size_t)LSEQ * DMOD * 4);
    float*    hid32     = (float*)alloc((size_t)LSEQ * DMOD * 4);
    _Float16* hn16      = (_Float16*)alloc((size_t)LSEQ * DMOD * 2);
    float*    xz32      = (float*)alloc((size_t)LSEQ * 2 * DI * 4);
    _Float16* xconv16   = (_Float16*)alloc((size_t)LSEQ * DI * 2);
    float*    dbc32     = (float*)alloc((size_t)LSEQ * DBCPAD * 4);
    _Float16* dbc16     = (_Float16*)alloc((size_t)LSEQ * RPAD * 2);
    float*    dtlin32   = (float*)alloc((size_t)LSEQ * DI * 4);
    _Float16* y16       = (_Float16*)alloc((size_t)LSEQ * DI * 2);
    _Float16* cat16     = (_Float16*)alloc((size_t)LSEQ * 2 * DMOD * 2);
    float*    proj32    = (float*)alloc((size_t)LSEQ * DMOD * 4);
    _Float16* proj16    = (_Float16*)alloc((size_t)LSEQ * DMOD * 2);
    (void)ws_size; (void)n_in; (void)in_sizes; (void)out_size;

    // ---- one-time weight prep ----
    gather_emb_kernel<<<LSEQ, 256, 0, stream>>>(ids, embedding, embF, embB);
    f32_to_f16_kernel<<<2048, 256, 0, stream>>>(embedding, emb16, VOC * DMOD);
    f32_to_f16_kernel<<<1024, 256, 0, stream>>>(lm_head, lmhead16, DMOD * DI);

    // ---- two directions ----
    for (int dir = 0; dir < 2; dir++) {
        const float* embIn = (dir == 0) ? embF : embB;
        for (int layer = 0; layer < NLAY; layer++) {
            const float* w_in   = P[dir][IN_PROJ] + (size_t)layer * 2 * DI * DMOD;
            const float* w_cw   = P[dir][CONV_W]  + (size_t)layer * DI * KCONV;
            const float* w_cb   = P[dir][CONV_B]  + (size_t)layer * DI;
            const float* w_xp   = P[dir][X_PROJ]  + (size_t)layer * DBCN * DI;
            const float* w_dtw  = P[dir][DT_W]    + (size_t)layer * DI * RRK;
            const float* w_dtb  = P[dir][DT_B]    + (size_t)layer * DI;
            const float* w_alog = P[dir][A_LOG]   + (size_t)layer * DI * SST;
            const float* w_dp   = P[dir][DP]      + (size_t)layer * DI;
            const float* w_out  = P[dir][OUT_PROJ]+ (size_t)layer * DMOD * DI;
            const float* w_nw   = P[dir][NORM_W]  + (size_t)layer * DMOD;

            f32_to_f16_kernel<<<2048, 256, 0, stream>>>(w_in, inproj16, 2 * DI * DMOD);
            pad_xproj_kernel<<<(DBCPAD * DI) / 256, 256, 0, stream>>>(w_xp, xproj16);
            pad_dtw_kernel<<<(DI * RPAD) / 256, 256, 0, stream>>>(w_dtw, dtw16);
            f32_to_f16_kernel<<<1024, 256, 0, stream>>>(w_out, outproj16, DMOD * DI);

            // residual update + rmsnorm -> hn16
            if (layer == 0)
                resnorm_kernel<<<LSEQ, 256, 0, stream>>>(embIn, nullptr, res32, w_nw,
                                                         hn16, DMOD, 0, 0);
            else
                resnorm_kernel<<<LSEQ, 256, 0, stream>>>(hid32, res32, res32, w_nw,
                                                         hn16, DMOD, 0, 0);

            // xz = hn @ in_proj^T : [2048, 3072]
            gemm_wmma_f16<<<dim3((2 * DI) / 128, LSEQ / 64), 256, 0, stream>>>(
                hn16, DMOD, inproj16, DMOD, xz32, 2 * DI, DMOD);

            // conv + silu -> xconv16 [2048, 1536]
            conv_silu_kernel<<<(LSEQ * DI) / 256, 256, 0, stream>>>(xz32, w_cw, w_cb, xconv16);

            // dbc = xconv @ x_proj^T : [2048, 128(pad of 80)]
            gemm_wmma_f16<<<dim3(DBCPAD / 128, LSEQ / 64), 256, 0, stream>>>(
                xconv16, DI, xproj16, DI, dbc32, DBCPAD, DI);

            make_dbc16_kernel<<<(LSEQ * RPAD) / 256, 256, 0, stream>>>(dbc32, dbc16);

            // dt_lin = dbc[:, :48] @ dt_w^T : [2048, 1536]
            gemm_wmma_f16<<<dim3(DI / 128, LSEQ / 64), 256, 0, stream>>>(
                dbc16, RPAD, dtw16, RPAD, dtlin32, DI, RPAD);

            // selective scan -> y16 [2048, 1536]
            scan_kernel<<<DI / 256, 256, 0, stream>>>(
                dtlin32, dbc32, xconv16, xz32, w_alog, w_dp, w_dtb, y16);

            // hidden = y @ out_proj^T : [2048, 768]
            gemm_wmma_f16<<<dim3(DMOD / 128, LSEQ / 64), 256, 0, stream>>>(
                y16, DI, outproj16, DI, hid32, DMOD, DI);
        }
        // final norm -> concat buffer (backward branch: flip rows, cols 768..1535)
        resnorm_kernel<<<LSEQ, 256, 0, stream>>>(
            hid32, res32, nullptr, P[dir][NORM_F],
            cat16, 2 * DMOD, dir * DMOD, dir ? 1 : 0);
    }

    // proj = cat @ lm_head^T : [2048, 768]
    gemm_wmma_f16<<<dim3(DMOD / 128, LSEQ / 64), 256, 0, stream>>>(
        cat16, 2 * DMOD, lmhead16, 2 * DMOD, proj32, DMOD, 2 * DMOD);
    f32_to_f16_kernel<<<1024, 256, 0, stream>>>(proj32, proj16, LSEQ * DMOD);

    // logits = proj @ embedding^T : [2048, 32000]  (dominant GEMM, ~100 GFLOP)
    gemm_wmma_f16<<<dim3(VOC / 128, LSEQ / 64), 256, 0, stream>>>(
        proj16, DMOD, emb16, DMOD, (float*)d_out, VOC, DMOD);
}
